// RotationTransformer_90529320665501
// MI455X (gfx1250) — matrix operand order repaired and checked by reading, not compiled, not run on hardware
//
#include <hip/hip_runtime.h>

typedef __attribute__((ext_vector_type(16))) _Float16 v16h;
typedef __attribute__((ext_vector_type(8)))  _Float16 v8h;
typedef __attribute__((ext_vector_type(8)))  float    v8f;

union AFrag { v16h v; _Float16 e[16]; v8h h[2]; };
union CFrag { v8f  v; float    e[8]; };
union H8    { v8h  v; _Float16 e[8]; };

__device__ __forceinline__ unsigned pack_f16x2(float lo, float hi) {
    union { _Float16 h[2]; unsigned u; } p;
    p.h[0] = (_Float16)lo; p.h[1] = (_Float16)hi;
    return p.u;
}

// ---------------------------------------------------------------------------
// Prepack conv2 weights into WMMA B-fragment layout with k-order
// k' = (kh*3+kw)*32 + ic  (matches channel-last implicit GEMM below).
// Layout: [ntile(4)][s(9)][lane(32)][8 x u32].
// Fragment element VGPR i holds (K=2i,2i+1) of the lane half:
//   ic_lo(i) = ((i<4) ? 2i : 2i+8) + 8*(lane>=16)
__global__ void pack_conv2_kernel(const float* __restrict__ w2,
                                  unsigned* __restrict__ w2pack) {
    int nt   = blockIdx.x / 9;        // 0..3
    int s    = blockIdx.x % 9;        // kh*3+kw
    int lane = threadIdx.x;           // 0..31
    int n    = nt * 16 + (lane & 15);
    int h    = (lane >> 4) & 1;
    unsigned* dst = w2pack + ((size_t)blockIdx.x * 32 + lane) * 8;
    #pragma unroll
    for (int i = 0; i < 8; ++i) {
        int ic = ((i < 4) ? 2 * i : 2 * i + 8) + 8 * h;
        dst[i] = pack_f16x2(w2[n * 288 + ic * 9 + s],
                            w2[n * 288 + (ic + 1) * 9 + s]);
    }
}

// Prepack fc1 weights: B[k][n] = fc1_w[n*9216 + k].
// Layout: [ntile(8)][ktile(288)][lane(32)][8 x u32].
__global__ void pack_fc1_kernel(const float* __restrict__ fc1w,
                                unsigned* __restrict__ fc1pack) {
    int g    = blockIdx.x * 256 + threadIdx.x;   // 0..73727
    int nt   = g / (288 * 32);
    int rem  = g % (288 * 32);
    int kt   = rem >> 5;
    int lane = rem & 31;
    int n     = nt * 16 + (lane & 15);
    int kbase = kt * 32 + ((lane >> 4) << 4);
    unsigned* dst = fc1pack + (size_t)g * 8;
    const float* row = fc1w + (size_t)n * 9216;
    #pragma unroll
    for (int i = 0; i < 8; ++i) {
        int k0 = kbase + 2 * i;
        dst[i] = pack_f16x2(row[k0], row[k0 + 1]);
    }
}

// ---------------------------------------------------------------------------
// Fused conv1 -> relu -> conv2 (implicit GEMM, 9 x K=32 WMMA steps over
// channel-last LDS activations, B fragments staged in LDS) -> relu -> pool.
// One workgroup (8 wave32) per image; one wave per 16-row M-tile x 4 N-tiles.
__global__ void __launch_bounds__(256) conv_pool_kernel(
    const float* __restrict__ x,
    const float* __restrict__ w1, const float* __restrict__ b1,
    const float* __restrict__ b2,
    const unsigned* __restrict__ w2pack,
    _Float16* __restrict__ pooled)
{
    __shared__ float    xs[784];
    __shared__ float    w1s[288];
    __shared__ float    b1s[32];
    __shared__ float    b2s[64];
    __shared__ unsigned w2s[9216];   // staged B fragments (36 KB)
    __shared__ _Float16 h1c[21632];  // [26][26][32] : (y*26+x)*32 + ic
    __shared__ _Float16 c2[36864];   // [64][576]    : oc*576 + oy*24 + ox

    const int tid = threadIdx.x;
    const int b   = blockIdx.x;
    const float* img = x + (size_t)b * 784;

    for (int i = tid; i < 784; i += 256) xs[i]  = img[i];
    for (int i = tid; i < 288; i += 256) w1s[i] = w1[i];
    if (tid < 32) b1s[tid] = b1[tid];
    if (tid < 64) b2s[tid] = b2[tid];
    // coalesced 16B copies: 9216 u32 = 2304 uint4
    for (int i = tid; i < 2304; i += 256)
        ((uint4*)w2s)[i] = ((const uint4*)w2pack)[i];
    __syncthreads();

    // conv1 + relu -> h1c (f16, channel-last)
    for (int idx = tid; idx < 21632; idx += 256) {
        int ic = idx & 31;           // output channel of conv1
        int p  = idx >> 5;           // spatial position 0..675
        int oy = p / 26, ox = p % 26;
        float s = b1s[ic];
        const float* wk = &w1s[ic * 9];
        #pragma unroll
        for (int kh = 0; kh < 3; ++kh)
            #pragma unroll
            for (int kw = 0; kw < 3; ++kw)
                s = fmaf(xs[(oy + kh) * 28 + ox + kw], wk[kh * 3 + kw], s);
        h1c[p * 32 + ic] = (_Float16)fmaxf(s, 0.0f);
    }
    __syncthreads();

    // conv2: D[576 x 64] = A[576 x 288] * B[288 x 64]
    // K-order (kh,kw,ic): per step, A slice is contiguous 32 channels.
    const int lane  = tid & 31;
    const int wave  = tid >> 5;
    const int lrow  = lane & 15;
    const int khalf = (lane >> 4) << 3;   // ic offset of lane half (0|8)
    const int mhalf = (lane >> 4) << 3;   // M offset for C/D regs (0|8)

    for (int mt = wave; mt < 36; mt += 8) {
        int m0 = mt * 16;
        int M  = m0 + lrow;               // A rows same in both lane halves
        int oy = M / 24, ox = M % 24;
        const int rowbase = (oy * 26 + ox) * 32;

        CFrag acc[4];
        #pragma unroll
        for (int nt = 0; nt < 4; ++nt) {
            float bias = b2s[nt * 16 + lrow];
            #pragma unroll
            for (int r = 0; r < 8; ++r) acc[nt].e[r] = bias;
        }

        #pragma unroll
        for (int kh = 0; kh < 3; ++kh) {
            #pragma unroll
            for (int kw = 0; kw < 3; ++kw) {
                const int s = kh * 3 + kw;
                const _Float16* src = &h1c[rowbase + (kh * 26 + kw) * 32 + khalf];
                AFrag a;
                a.h[0] = *(const v8h*)(src);        // ic khalf .. khalf+7
                a.h[1] = *(const v8h*)(src + 16);   // ic khalf+16 .. +23
                #pragma unroll
                for (int nt = 0; nt < 4; ++nt) {
                    const v16h bb = *(const v16h*)&w2s[((nt * 9 + s) * 32 + lane) * 8];
                    acc[nt].v = __builtin_amdgcn_wmma_f32_16x16x32_f16(
                        false, a.v, false, bb, (short)0, acc[nt].v, false, false);
                }
            }
        }

        #pragma unroll
        for (int nt = 0; nt < 4; ++nt) {
            int n = nt * 16 + lrow;
            H8 st;
            #pragma unroll
            for (int r = 0; r < 8; ++r)
                st.e[r] = (_Float16)fmaxf(acc[nt].e[r], 0.0f);
            *(v8h*)&c2[n * 576 + m0 + mhalf] = st.v;   // M contiguous over r
        }
    }
    __syncthreads();

    // 2x2 maxpool -> pooled row [9216], layout oc*144 + py*12 + px
    _Float16* prow = pooled + (size_t)b * 9216;
    for (int idx = tid; idx < 9216; idx += 256) {
        int oc = idx / 144;
        int q  = idx % 144;
        int py = q / 12, px = q % 12;
        const _Float16* base = &c2[oc * 576 + (py * 2) * 24 + px * 2];
        float v00 = (float)base[0],  v01 = (float)base[1];
        float v10 = (float)base[24], v11 = (float)base[25];
        prow[idx] = (_Float16)fmaxf(fmaxf(v00, v01), fmaxf(v10, v11));
    }
}

// ---------------------------------------------------------------------------
// fc1: [8192 x 9216] * [9216 x 128] + bias, relu -> hfc1t (f16, transposed
// [128][8192] so each lane stores 8 contiguous f16).
// One wave per 16-row M-tile, covering all 8 N-tiles (64 f32 accumulators).
__global__ void __launch_bounds__(256) fc1_gemm_kernel(
    const _Float16* __restrict__ pooled,
    const unsigned* __restrict__ fc1pack,
    const float* __restrict__ fc1b,
    _Float16* __restrict__ hfc1t)
{
    const int lane  = threadIdx.x & 31;
    const int wave  = threadIdx.x >> 5;
    const int mt    = blockIdx.x * 8 + wave;   // 0..511
    const int m0    = mt * 16;
    const int lrow  = lane & 15;
    const int koff  = (lane >> 4) << 4;
    const int mhalf = (lane >> 4) << 3;

    const _Float16* arow = pooled + (size_t)(m0 + lrow) * 9216;

    CFrag acc[8];
    #pragma unroll
    for (int nt = 0; nt < 8; ++nt)
        #pragma unroll
        for (int r = 0; r < 8; ++r) acc[nt].e[r] = 0.0f;

    for (int kt = 0; kt < 288; ++kt) {
        int k0 = kt * 32 + koff;
        AFrag a;
        a.h[0] = *(const v8h*)(arow + k0);        // K j..j+7
        a.h[1] = *(const v8h*)(arow + k0 + 8);    // K j+8..j+15
        #pragma unroll
        for (int nt = 0; nt < 8; ++nt) {
            const v16h bb = *(const v16h*)(fc1pack +
                ((size_t)((nt * 288 + kt) * 32 + lane) * 8));
            acc[nt].v = __builtin_amdgcn_wmma_f32_16x16x32_f16(
                false, a.v, false, bb, (short)0, acc[nt].v, false, false);
        }
    }

    #pragma unroll
    for (int nt = 0; nt < 8; ++nt) {
        int n = nt * 16 + lrow;
        float bias = fc1b[n];
        H8 st;
        #pragma unroll
        for (int r = 0; r < 8; ++r)
            st.e[r] = (_Float16)fmaxf(acc[nt].e[r] + bias, 0.0f);
        *(v8h*)&hfc1t[(size_t)n * 8192 + m0 + mhalf] = st.v;
    }
}

// ---------------------------------------------------------------------------
// fc2 + argmax routing + bilinear rotation. One workgroup per image.
__global__ void __launch_bounds__(256) head_rotate_kernel(
    const float* __restrict__ x,
    const _Float16* __restrict__ hfc1t,
    const float* __restrict__ fc2w,
    const float* __restrict__ fc2b,
    float* __restrict__ out)
{
    __shared__ float hrow[128];
    __shared__ float logits[8];
    __shared__ float xs[784];
    __shared__ float s_ct, s_st;

    const int tid = threadIdx.x;
    const int b   = blockIdx.x;

    if (tid < 128) hrow[tid] = (float)hfc1t[(size_t)tid * 8192 + b];
    for (int i = tid; i < 784; i += 256) xs[i] = x[(size_t)b * 784 + i];
    __syncthreads();

    if (tid < 8) {
        float s = fc2b[tid];
        const float* w = fc2w + tid * 128;
        #pragma unroll 4
        for (int k = 0; k < 128; ++k) s = fmaf(hrow[k], w[k], s);
        logits[tid] = s;
    }
    __syncthreads();

    if (tid == 0) {
        float best = logits[0]; int bi = 0;
        #pragma unroll
        for (int j = 1; j < 8; ++j) if (logits[j] > best) { best = logits[j]; bi = j; }
        float ang = (float)bi * (15.0f * 3.14159265358979323846f / 180.0f);
        s_ct = cosf(ang);
        s_st = sinf(ang);
        out[(size_t)8192 * 784 + b] = (bi != 0) ? 1.0f : 0.0f;
    }
    __syncthreads();

    const float ct = s_ct, st = s_st;
    const float c = 13.5f;   // (28-1)/2
    for (int p = tid; p < 784; p += 256) {
        int yi = p / 28, xi = p % 28;
        float dx = (float)xi - c, dy = (float)yi - c;
        float sx =  ct * dx + st * dy + c;
        float sy = -st * dx + ct * dy + c;
        float x0f = floorf(sx), y0f = floorf(sy);
        float wx = sx - x0f, wy = sy - y0f;
        int ix0 = (int)x0f, iy0 = (int)y0f;
        bool xa = (ix0 >= 0) & (ix0 < 28);
        bool xb = (ix0 + 1 >= 0) & (ix0 + 1 < 28);
        bool ya = (iy0 >= 0) & (iy0 < 28);
        bool yb = (iy0 + 1 >= 0) & (iy0 + 1 < 28);
        float v00 = (xa && ya) ? xs[iy0 * 28 + ix0]           : 0.0f;
        float v01 = (xb && ya) ? xs[iy0 * 28 + ix0 + 1]       : 0.0f;
        float v10 = (xa && yb) ? xs[(iy0 + 1) * 28 + ix0]     : 0.0f;
        float v11 = (xb && yb) ? xs[(iy0 + 1) * 28 + ix0 + 1] : 0.0f;
        out[(size_t)b * 784 + p] =
            v00 * (1.0f - wy) * (1.0f - wx) + v01 * (1.0f - wy) * wx +
            v10 * wy * (1.0f - wx)          + v11 * wy * wx;
    }
}

// ---------------------------------------------------------------------------
extern "C" void kernel_launch(void* const* d_in, const int* in_sizes, int n_in,
                              void* d_out, int out_size, void* d_ws, size_t ws_size,
                              hipStream_t stream) {
    (void)in_sizes; (void)n_in; (void)out_size; (void)ws_size;
    const float* x    = (const float*)d_in[0];
    const float* w1   = (const float*)d_in[1];
    const float* b1   = (const float*)d_in[2];
    const float* w2   = (const float*)d_in[3];
    const float* b2   = (const float*)d_in[4];
    const float* fc1w = (const float*)d_in[5];
    const float* fc1b = (const float*)d_in[6];
    const float* fc2w = (const float*)d_in[7];
    const float* fc2b = (const float*)d_in[8];
    float* out = (float*)d_out;

    // Workspace layout (bytes):
    //   [0)           w2pack   : 4*9*32*8*4   = 36864
    //   [36864)       fc1pack  : 8*288*32*8*4 = 2359296
    //   [2396160)     pooled   : 8192*9216*2  = 150994944  (f16)
    //   [153391104)   hfc1t    : 128*8192*2   = 2097152    (f16, transposed)
    char* ws = (char*)d_ws;
    unsigned* w2pack  = (unsigned*)(ws);
    unsigned* fc1pack = (unsigned*)(ws + 36864);
    _Float16* pooled  = (_Float16*)(ws + 2396160);
    _Float16* hfc1t   = (_Float16*)(ws + 153391104);

    pack_conv2_kernel<<<36, 32, 0, stream>>>(w2, w2pack);
    pack_fc1_kernel<<<288, 256, 0, stream>>>(fc1w, fc1pack);
    conv_pool_kernel<<<8192, 256, 0, stream>>>(x, w1, b1, b2, w2pack, pooled);
    fc1_gemm_kernel<<<64, 256, 0, stream>>>(pooled, fc1pack, fc1b, hfc1t);
    head_rotate_kernel<<<8192, 256, 0, stream>>>(x, hfc1t, fc2w, fc2b, out);
}
